// PiNet_15453292331024
// MI455X (gfx1250) — matrix operand order repaired and checked by reading, not compiled, not run on hardware
//
#include <hip/hip_runtime.h>
#include <hip/hip_bf16.h>

// ---------------- problem constants (match reference) ----------------
constexpr int G     = 64;
constexpr int NPG   = 1024;
constexpr int N     = G * NPG;      // 65536
constexpr int DEG   = 32;
constexpr int E     = N * DEG;      // 2097152
constexpr int F_IN  = 128;
constexpr int D0    = 64;
constexpr int SKIP2 = 128;

#define DEV __device__ __forceinline__

// ---------------- CDNA5 WMMA types ----------------
typedef __bf16        v16bf __attribute__((ext_vector_type(16)));
typedef float         v8f   __attribute__((ext_vector_type(8)));
typedef unsigned int  u32x4 __attribute__((ext_vector_type(4)));

DEV v8f wmma_bf16(v16bf a, v16bf b, v8f c) {
  // D(f32,16x16) = A(bf16,16x32) * B(bf16,32x16) + C
  return __builtin_amdgcn_wmma_f32_16x16x32_bf16(
      /*neg_a=*/false, a, /*neg_b=*/false, b,
      /*c_mod=*/(short)0, c, /*reuse_a=*/false, /*reuse_b=*/false);
}

// Fragment-native packed load: tile = 32 lanes x 16 halves, lane-major.
// Each lane: two contiguous 16B loads.
DEV v16bf load_frag(const __bf16* P, size_t tileIdx, int lane) {
  const __bf16* p = P + (tileIdx * 32 + (size_t)lane) * 16;
  union { v16bf v; u32x4 q[2]; } u;
  u.q[0] = *(const u32x4*)(p);
  u.q[1] = *(const u32x4*)(p + 8);
  return u.v;
}

// A-matrix 16x32 bf16 fragment from a row-major bf16 activation buffer.
// Layout: lane L (0..15) row M=L holds K 0..15 (elems 0..15);
// lanes 16..31 hold K 16..31. Two 16B loads per lane.
DEV v16bf load_a_row(const __bf16* A, int row0, int ld, int kbase, int lane) {
  int m     = lane & 15;
  int khalf = (lane >> 4) * 16;
  const __bf16* p = A + (size_t)(row0 + m) * ld + kbase + khalf;
  union { v16bf v; u32x4 q[2]; } u;
  u.q[0] = *(const u32x4*)(p);        // K = kbase+khalf .. +7
  u.q[1] = *(const u32x4*)(p + 8);    // K = kbase+khalf+8 .. +15
  return u.v;
}

DEV v8f vzero() { v8f z = {0.f,0.f,0.f,0.f,0.f,0.f,0.f,0.f}; return z; }

// ---------------- small utility kernels ----------------
__global__ void k_deg_init(float* deg) {
  int i = blockIdx.x * blockDim.x + threadIdx.x;
  if (i < N) deg[i] = 1.0f;                       // self loop
}
__global__ void k_deg_count(const int* __restrict__ dst, float* deg) {
  int e = blockIdx.x * blockDim.x + threadIdx.x;
  if (e < E) atomicAdd(&deg[dst[e]], 1.0f);
}
__global__ void k_rsqrt(float* d) {
  int i = blockIdx.x * blockDim.x + threadIdx.x;
  if (i < N) d[i] = rsqrtf(d[i]);
}
__global__ void k_cvt_bf16(const float* __restrict__ x, __bf16* __restrict__ y, int n) {
  int i = blockIdx.x * blockDim.x + threadIdx.x;
  if (i < n) y[i] = (__bf16)x[i];
}

// Pack a f32 weight [K x 64] into B-fragment-native bf16 layout:
//   P[((kt*4 + nt)*32 + lane)*16 + i] = W[(kt*32 + (lane>>4)*16 + i)*64 + nt*16 + (lane&15)]
__global__ void k_pack_w(const float* __restrict__ W, __bf16* __restrict__ P, int K) {
  int t = blockIdx.x * blockDim.x + threadIdx.x;
  if (t >= K * 64) return;
  int i    = t & 15;
  int lane = (t >> 4) & 31;
  int nt   = (t >> 9) & 3;
  int kt   = t >> 11;
  int k = kt * 32 + (lane >> 4) * 16 + i;
  int n = nt * 16 + (lane & 15);
  P[t] = (__bf16)W[(size_t)k * 64 + n];
}

// out = hw * dinv^2 + bias   (self-loop term + bias, both branches)
__global__ void k_agg_init(const float* __restrict__ hwa, const float* __restrict__ hwx,
                           const float* __restrict__ dinv,
                           const float* __restrict__ ba, const float* __restrict__ bx,
                           float* __restrict__ ao, float* __restrict__ xo) {
  int idx = blockIdx.x * blockDim.x + threadIdx.x;
  if (idx >= N * D0) return;
  int n = idx >> 6, f = idx & 63;
  float w = dinv[n] * dinv[n];
  ao[idx] = hwa[idx] * w + ba[f];
  xo[idx] = hwx[idx] * w + bx[f];
}

// scatter-add over edges, both branches fused; thread = (edge, 16-feat chunk)
__global__ void k_agg_edges(const int* __restrict__ src, const int* __restrict__ dst,
                            const float* __restrict__ dinv,
                            const float* __restrict__ hwa, const float* __restrict__ hwx,
                            float* __restrict__ ao, float* __restrict__ xo) {
  int t = blockIdx.x * blockDim.x + threadIdx.x;
  if (t >= E * 4) return;
  int e  = t >> 2;
  int f0 = (t & 3) * 16;
  int s = src[e], d = dst[e];
  float w = dinv[s] * dinv[d];
  const float4* pa = (const float4*)(hwa + (size_t)s * D0 + f0);
  const float4* px = (const float4*)(hwx + (size_t)s * D0 + f0);
  float* oa = ao + (size_t)d * D0 + f0;
  float* ox = xo + (size_t)d * D0 + f0;
#pragma unroll
  for (int j = 0; j < 4; j++) {
    float4 va = pa[j], vx = px[j];
    atomicAdd(oa + j*4 + 0, va.x * w); atomicAdd(oa + j*4 + 1, va.y * w);
    atomicAdd(oa + j*4 + 2, va.z * w); atomicAdd(oa + j*4 + 3, va.w * w);
    atomicAdd(ox + j*4 + 0, vx.x * w); atomicAdd(ox + j*4 + 1, vx.y * w);
    atomicAdd(ox + j*4 + 2, vx.z * w); atomicAdd(ox + j*4 + 3, vx.w * w);
  }
}

// relu + bf16 for next GEMM (f32 relu'd copies not needed downstream)
__global__ void k_finish1(const float* __restrict__ a1pre, const float* __restrict__ x1pre,
                          __bf16* __restrict__ a1b, __bf16* __restrict__ x1b) {
  int idx = blockIdx.x * blockDim.x + threadIdx.x;
  if (idx >= N * D0) return;
  a1b[idx] = (__bf16)fmaxf(a1pre[idx], 0.0f);
  x1b[idx] = (__bf16)fmaxf(x1pre[idx], 0.0f);
}

// build x2cat = [relu(x2pre) | x1L] directly in bmm B-fragment-packed layout:
//   for node, col c: tile = (node>>5)*8 + (c>>4); lane = (node&16)?16:0 + (c&15)... see below
__global__ void k_finish2(const float* __restrict__ x2pre, const __bf16* __restrict__ xLb,
                          __bf16* __restrict__ xcatP) {
  int idx = blockIdx.x * blockDim.x + threadIdx.x;
  if (idx >= N * D0) return;
  int node = idx >> 6, f = idx & 63;
  int kb = node >> 5, kk = node & 31;
  int lanehi = (kk >> 4) * 16;    // lanes 0-15: K 0..15 ; 16-31: K 16..31
  int ii = kk & 15;
  // left half: col f  = relu(x2pre)
  {
    int lane = lanehi + (f & 15), nt = f >> 4;
    xcatP[(((size_t)kb * 8 + nt) * 32 + lane) * 16 + ii] = (__bf16)fmaxf(x2pre[idx], 0.0f);
  }
  // right half: col 64+f = x1L (already bf16)
  {
    int c = 64 + f;
    int lane = lanehi + (c & 15), nt = c >> 4;
    xcatP[(((size_t)kb * 8 + nt) * 32 + lane) * 16 + ii] = xLb[idx];
  }
}

// ---------------- WMMA GEMM kernels ----------------
// Stage 1: hwa = x@Wa1, hwx = x@Wx1.   block = 4 waves x 16 rows = 64 rows
__global__ void k_gemm_xw(const __bf16* __restrict__ xb,
                          const __bf16* __restrict__ PWa, const __bf16* __restrict__ PWx,
                          float* __restrict__ outa, float* __restrict__ outx) {
  int lane = threadIdx.x & 31, wave = threadIdx.x >> 5;
  int row0 = blockIdx.x * 64 + wave * 16;
  v8f accA[4], accX[4];
#pragma unroll
  for (int i = 0; i < 4; i++) { accA[i] = vzero(); accX[i] = vzero(); }
#pragma unroll
  for (int kt = 0; kt < 4; kt++) {                   // K = 128
    v16bf a = load_a_row(xb, row0, F_IN, kt * 32, lane);
#pragma unroll
    for (int nt = 0; nt < 4; nt++) {
      accA[nt] = wmma_bf16(a, load_frag(PWa, kt * 4 + nt, lane), accA[nt]);
      accX[nt] = wmma_bf16(a, load_frag(PWx, kt * 4 + nt, lane), accX[nt]);
    }
  }
  int mrow = row0 + (lane >> 4) * 8, ncol = lane & 15;
#pragma unroll
  for (int nt = 0; nt < 4; nt++)
#pragma unroll
    for (int r = 0; r < 8; r++) {
      outa[(size_t)(mrow + r) * D0 + nt * 16 + ncol] = accA[nt][r];
      outx[(size_t)(mrow + r) * D0 + nt * 16 + ncol] = accX[nt][r];
    }
}

// Stage 2: concat-K linear: a1L = [a1|x]@Wla + bla ; x1L = [x1|x]@Wlx + blx (K=192)
__global__ void k_gemm_lin(const __bf16* __restrict__ a1b, const __bf16* __restrict__ x1b,
                           const __bf16* __restrict__ xb,
                           const __bf16* __restrict__ PWla, const __bf16* __restrict__ PWlx,
                           const float* __restrict__ bla, const float* __restrict__ blx,
                           float* __restrict__ aLf, __bf16* __restrict__ aLb,
                           __bf16* __restrict__ xLb) {
  int lane = threadIdx.x & 31, wave = threadIdx.x >> 5;
  int row0 = blockIdx.x * 64 + wave * 16;
  v8f accA[4], accX[4];
#pragma unroll
  for (int i = 0; i < 4; i++) { accA[i] = vzero(); accX[i] = vzero(); }
#pragma unroll
  for (int kt = 0; kt < 2; kt++) {                   // first 64 K: a1 / x1
    v16bf aA = load_a_row(a1b, row0, D0, kt * 32, lane);
    v16bf aX = load_a_row(x1b, row0, D0, kt * 32, lane);
#pragma unroll
    for (int nt = 0; nt < 4; nt++) {
      accA[nt] = wmma_bf16(aA, load_frag(PWla, kt * 4 + nt, lane), accA[nt]);
      accX[nt] = wmma_bf16(aX, load_frag(PWlx, kt * 4 + nt, lane), accX[nt]);
    }
  }
#pragma unroll
  for (int kt = 2; kt < 6; kt++) {                   // last 128 K: shared x
    v16bf aS = load_a_row(xb, row0, F_IN, (kt - 2) * 32, lane);
#pragma unroll
    for (int nt = 0; nt < 4; nt++) {
      accA[nt] = wmma_bf16(aS, load_frag(PWla, kt * 4 + nt, lane), accA[nt]);
      accX[nt] = wmma_bf16(aS, load_frag(PWlx, kt * 4 + nt, lane), accX[nt]);
    }
  }
  int mrow = row0 + (lane >> 4) * 8, ncol = lane & 15;
#pragma unroll
  for (int nt = 0; nt < 4; nt++) {
    int n = nt * 16 + ncol;
    float bA = bla[n], bX = blx[n];
#pragma unroll
    for (int r = 0; r < 8; r++) {
      float va = accA[nt][r] + bA;
      float vx = accX[nt][r] + bX;
      size_t o = (size_t)(mrow + r) * D0 + n;
      aLf[o] = va; aLb[o] = (__bf16)va; xLb[o] = (__bf16)vx;
    }
  }
}

// Stage 3: hw2a = a1L@Wa2, hw2x = x1L@Wx2 (K=64, no bias)
__global__ void k_gemm_h2(const __bf16* __restrict__ aLb, const __bf16* __restrict__ xLb,
                          const __bf16* __restrict__ PWa2, const __bf16* __restrict__ PWx2,
                          float* __restrict__ outa, float* __restrict__ outx) {
  int lane = threadIdx.x & 31, wave = threadIdx.x >> 5;
  int row0 = blockIdx.x * 64 + wave * 16;
  v8f accA[4], accX[4];
#pragma unroll
  for (int i = 0; i < 4; i++) { accA[i] = vzero(); accX[i] = vzero(); }
#pragma unroll
  for (int kt = 0; kt < 2; kt++) {
    v16bf aA = load_a_row(aLb, row0, D0, kt * 32, lane);
    v16bf aX = load_a_row(xLb, row0, D0, kt * 32, lane);
#pragma unroll
    for (int nt = 0; nt < 4; nt++) {
      accA[nt] = wmma_bf16(aA, load_frag(PWa2, kt * 4 + nt, lane), accA[nt]);
      accX[nt] = wmma_bf16(aX, load_frag(PWx2, kt * 4 + nt, lane), accX[nt]);
    }
  }
  int mrow = row0 + (lane >> 4) * 8, ncol = lane & 15;
#pragma unroll
  for (int nt = 0; nt < 4; nt++)
#pragma unroll
    for (int r = 0; r < 8; r++) {
      outa[(size_t)(mrow + r) * D0 + nt * 16 + ncol] = accA[nt][r];
      outx[(size_t)(mrow + r) * D0 + nt * 16 + ncol] = accX[nt][r];
    }
}

// per-graph, per-column softmax over [a2 | a1L]; writes A-fragment-packed bf16.
// 1024 threads: f = col, j = 8-way node partition; LDS tree reduction.
__global__ void k_softmax(const float* __restrict__ a2pre, const float* __restrict__ a1L,
                          __bf16* __restrict__ a_smP) {
  int g = blockIdx.x, t = threadIdx.x;
  int f = t & 127, j = t >> 7;                 // j in 0..7
  int base = g * NPG;
  __shared__ float red[8][128];

  float m = -INFINITY;
  for (int i = j * 128; i < (j + 1) * 128; i++) {
    float v = (f < 64) ? a2pre[(size_t)(base + i) * D0 + f]
                       : a1L[(size_t)(base + i) * D0 + (f - 64)];
    m = fmaxf(m, v);
  }
  red[j][f] = m;
  __syncthreads();
  if (j == 0) {
    float mm = red[0][f];
#pragma unroll
    for (int q = 1; q < 8; q++) mm = fmaxf(mm, red[q][f]);
    red[0][f] = mm;
  }
  __syncthreads();
  m = red[0][f];
  __syncthreads();

  float s = 0.f;
  for (int i = j * 128; i < (j + 1) * 128; i++) {
    float v = (f < 64) ? a2pre[(size_t)(base + i) * D0 + f]
                       : a1L[(size_t)(base + i) * D0 + (f - 64)];
    s += __expf(v - m);
  }
  red[j][f] = s;
  __syncthreads();
  if (j == 0) {
    float ss = red[0][f];
#pragma unroll
    for (int q = 1; q < 8; q++) ss += red[q][f];
    red[0][f] = ss;
  }
  __syncthreads();
  float inv = 1.0f / red[0][f];

  int ft = f >> 4, mcol = f & 15;
  for (int i = j * 128; i < (j + 1) * 128; i++) {
    float v = (f < 64) ? a2pre[(size_t)(base + i) * D0 + f]
                       : a1L[(size_t)(base + i) * D0 + (f - 64)];
    float e = __expf(v - m) * inv;
    int node = base + i;
    int kb = node >> 5, kk = node & 31;
    int lane = mcol + 16 * (kk >> 4);
    int ii = kk & 15;
    a_smP[(((size_t)kb * 8 + ft) * 32 + lane) * 16 + ii] = (__bf16)e;
  }
}

// per-graph bilinear pooling: P = A^T (128x1024) @ X (1024x128) on WMMA.
// Both operands pre-packed in fragment layout: each fragment = two b128 loads.
// block = 1 graph, 8 waves; wave ft owns 16x128 strip of P; K=1024 in 32-steps.
__global__ void k_bmm(const __bf16* __restrict__ a_smP, const __bf16* __restrict__ xcatP,
                      float* __restrict__ flat) {
  int g = blockIdx.x;
  int lane = threadIdx.x & 31, ft = threadIdx.x >> 5;
  v8f acc[8];
#pragma unroll
  for (int i = 0; i < 8; i++) acc[i] = vzero();
  for (int kb2 = 0; kb2 < NPG / 32; kb2++) {
    size_t kb = (size_t)g * (NPG / 32) + kb2;
    if (kb2 + 1 < NPG / 32) {
      __builtin_prefetch(a_smP + (((kb + 1) * 8 + ft) * 32 + lane) * 16, 0, 3);
      __builtin_prefetch(xcatP + (((kb + 1) * 8 + ft) * 32 + lane) * 16, 0, 3);
    }
    v16bf a = load_frag(a_smP, kb * 8 + ft, lane);
#pragma unroll
    for (int nt = 0; nt < 8; nt++)
      acc[nt] = wmma_bf16(a, load_frag(xcatP, kb * 8 + nt, lane), acc[nt]);
  }
  float* P = flat + (size_t)g * SKIP2 * SKIP2;
  int mrow = ft * 16 + (lane >> 4) * 8, ncol = lane & 15;
#pragma unroll
  for (int nt = 0; nt < 8; nt++)
#pragma unroll
    for (int r = 0; r < 8; r++)
      P[(size_t)(mrow + r) * SKIP2 + nt * 16 + ncol] = acc[nt][r];
}

// batch_out = flat @ W2 + b2   (2 classes per graph, block reduction)
__global__ void k_head(const float* __restrict__ flat, const float* __restrict__ W2,
                       const float* __restrict__ b2, float* __restrict__ out) {
  int g = blockIdx.x, t = threadIdx.x;        // 256 threads
  float s0 = 0.f, s1 = 0.f;
  const float* F = flat + (size_t)g * 16384;
  for (int i = t; i < 16384; i += 256) {
    float v = F[i];
    s0 += v * W2[(size_t)i * 2 + 0];
    s1 += v * W2[(size_t)i * 2 + 1];
  }
  __shared__ float r0[256], r1[256];
  r0[t] = s0; r1[t] = s1;
  __syncthreads();
  for (int o = 128; o > 0; o >>= 1) {
    if (t < o) { r0[t] += r0[t + o]; r1[t] += r1[t + o]; }
    __syncthreads();
  }
  if (t == 0) {
    out[g * 2 + 0] = r0[0] + b2[0];
    out[g * 2 + 1] = r1[0] + b2[1];
  }
}

// ---------------- launcher ----------------
extern "C" void kernel_launch(void* const* d_in, const int* in_sizes, int n_in,
                              void* d_out, int out_size, void* d_ws, size_t ws_size,
                              hipStream_t stream) {
  const float* x   = (const float*)d_in[0];
  const int*   ei  = (const int*)d_in[1];
  const int*   src = ei;
  const int*   dst = ei + E;
  const float* Wa1 = (const float*)d_in[3];
  const float* ba1 = (const float*)d_in[4];
  const float* Wx1 = (const float*)d_in[5];
  const float* bx1 = (const float*)d_in[6];
  const float* Wla = (const float*)d_in[7];
  const float* bla = (const float*)d_in[8];
  const float* Wlx = (const float*)d_in[9];
  const float* blx = (const float*)d_in[10];
  const float* Wa2 = (const float*)d_in[11];
  const float* ba2 = (const float*)d_in[12];
  const float* Wx2 = (const float*)d_in[13];
  const float* bx2 = (const float*)d_in[14];
  const float* W2  = (const float*)d_in[15];
  const float* b2  = (const float*)d_in[16];

  float* out  = (float*)d_out;
  float* flat = out + G * 2;                    // tuple: [batch_out | flat]

  // workspace carve-out (~176 MB), 256B-aligned regions
  char* ws = (char*)d_ws;
  size_t off = 0;
  auto take = [&](size_t bytes) -> char* {
    char* p = ws + off;
    off = (off + bytes + 255) & ~(size_t)255;
    return p;
  };
  float*  dinv  = (float*) take((size_t)N * 4);
  __bf16* xb    = (__bf16*)take((size_t)N * F_IN * 2);
  float*  hwa   = (float*) take((size_t)N * D0 * 4);   // also hw2a
  float*  hwx   = (float*) take((size_t)N * D0 * 4);   // also hw2x
  float*  a1pre = (float*) take((size_t)N * D0 * 4);   // later holds a1L (f32)
  float*  x1pre = (float*) take((size_t)N * D0 * 4);
  __bf16* a1b   = (__bf16*)take((size_t)N * D0 * 2);
  __bf16* x1b   = (__bf16*)take((size_t)N * D0 * 2);
  __bf16* aLb   = (__bf16*)take((size_t)N * D0 * 2);
  __bf16* xLb   = (__bf16*)take((size_t)N * D0 * 2);
  float*  a2pre = (float*) take((size_t)N * D0 * 4);
  float*  x2pre = (float*) take((size_t)N * D0 * 4);
  __bf16* a_smP = (__bf16*)take((size_t)N * SKIP2 * 2);  // A-fragment packed
  __bf16* xcatP = (__bf16*)take((size_t)N * SKIP2 * 2);  // B-fragment packed
  // packed bf16 weights (tiny)
  __bf16* PWa1 = (__bf16*)take((size_t)F_IN * D0 * 2);
  __bf16* PWx1 = (__bf16*)take((size_t)F_IN * D0 * 2);
  __bf16* PWla = (__bf16*)take((size_t)(D0 + F_IN) * D0 * 2);
  __bf16* PWlx = (__bf16*)take((size_t)(D0 + F_IN) * D0 * 2);
  __bf16* PWa2 = (__bf16*)take((size_t)D0 * D0 * 2);
  __bf16* PWx2 = (__bf16*)take((size_t)D0 * D0 * 2);
  (void)ws_size; (void)in_sizes; (void)n_in; (void)out_size;

  // degrees / symmetric norm; input conversion; weight packing
  k_deg_init <<<N / 256, 256, 0, stream>>>(dinv);
  k_deg_count<<<E / 256, 256, 0, stream>>>(dst, dinv);
  k_rsqrt    <<<N / 256, 256, 0, stream>>>(dinv);
  k_cvt_bf16 <<<(N * F_IN) / 256, 256, 0, stream>>>(x, xb, N * F_IN);
  k_pack_w   <<<(F_IN * 64) / 256, 256, 0, stream>>>(Wa1, PWa1, F_IN);
  k_pack_w   <<<(F_IN * 64) / 256, 256, 0, stream>>>(Wx1, PWx1, F_IN);
  k_pack_w   <<<((D0 + F_IN) * 64) / 256, 256, 0, stream>>>(Wla, PWla, D0 + F_IN);
  k_pack_w   <<<((D0 + F_IN) * 64) / 256, 256, 0, stream>>>(Wlx, PWlx, D0 + F_IN);
  k_pack_w   <<<(D0 * 64) / 256, 256, 0, stream>>>(Wa2, PWa2, D0);
  k_pack_w   <<<(D0 * 64) / 256, 256, 0, stream>>>(Wx2, PWx2, D0);

  // conv1 transforms + aggregation + relu
  k_gemm_xw  <<<N / 64, 128, 0, stream>>>(xb, PWa1, PWx1, hwa, hwx);
  k_agg_init <<<(N * D0) / 256, 256, 0, stream>>>(hwa, hwx, dinv, ba1, bx1, a1pre, x1pre);
  k_agg_edges<<<(E * 4) / 256, 256, 0, stream>>>(src, dst, dinv, hwa, hwx, a1pre, x1pre);
  k_finish1  <<<(N * D0) / 256, 256, 0, stream>>>(a1pre, x1pre, a1b, x1b);

  // skip-concat linears (K=192 concat GEMM); a1L overwrites a1pre buffer
  k_gemm_lin <<<N / 64, 128, 0, stream>>>(a1b, x1b, xb, PWla, PWlx, bla, blx,
                                          a1pre, aLb, xLb);

  // conv2 transforms + aggregation
  k_gemm_h2  <<<N / 64, 128, 0, stream>>>(aLb, xLb, PWa2, PWx2, hwa, hwx);
  k_agg_init <<<(N * D0) / 256, 256, 0, stream>>>(hwa, hwx, dinv, ba2, bx2, a2pre, x2pre);
  k_agg_edges<<<(E * 4) / 256, 256, 0, stream>>>(src, dst, dinv, hwa, hwx, a2pre, x2pre);
  k_finish2  <<<(N * D0) / 256, 256, 0, stream>>>(x2pre, xLb, xcatP);

  // segment softmax over [a2 | a1L] (packed write), per-graph A^T X on WMMA, head
  k_softmax  <<<G, 1024, 0, stream>>>(a2pre, a1pre, a_smP);
  k_bmm      <<<G, 256, 0, stream>>>(a_smP, xcatP, flat);
  k_head     <<<G, 256, 0, stream>>>(flat, W2, b2, out);
}
